// Net_3959959847510
// MI455X (gfx1250) — matrix operand order repaired and checked by reading, not compiled, not run on hardware
//
#include <hip/hip_runtime.h>

#define DIM 128
#define NN 256
#define NE 8
#define NG 4
#define ITERS 10
#define MAX_TILES 32

typedef __attribute__((ext_vector_type(16))) _Float16 v16h;
typedef __attribute__((ext_vector_type(4)))  _Float16 v4h;
typedef __attribute__((ext_vector_type(8)))  float    v8f;

// ---------------- workspace layout (bytes) ----------------
#define WS_NT       0            // 256*128 f32        = 131072
#define WS_WH       131072       // 8*128*128 f16      = 262144
#define WS_MSG255   393216       // 8*128 f32          =   4096
#define WS_CLO      397312       // 4*256 i32          =   4096
#define WS_CHI      401408       // 4*256 i32          =   4096
#define WS_NODELIST 405504       // 32*16 i32          =   2048
#define WS_TILETYPE 407552       // 32 i32             =    128
#define WS_NTILES   407680       // 1 i32
#define WS_SCORES   407808       // 32 f32

// ---------------- LDS layout (bytes) ----------------
#define LDS_EMBH    0                          // [256][128] f16 = 65536
#define LDS_MSGS    65536                      // [256][128] f32 = 131072
#define LDS_RED     (65536 + 131072)           // float[256]     = 1024
#define LDS_CLO     (65536 + 131072 + 1024)    // int[256]       = 1024
#define LDS_CHI     (65536 + 131072 + 2048)    // int[256]       = 1024
#define LDS_BYTES   (65536 + 131072 + 3072)    // 199680

// ---------------- prep kernels ----------------

__global__ void k_node_term(const int* __restrict__ data,
                            const float* __restrict__ vecs,
                            const float* __restrict__ w,
                            const float* __restrict__ b,
                            float* __restrict__ nt) {
    int n = blockIdx.x, d = threadIdx.x;
    nt[n * DIM + d] = vecs[(size_t)data[n] * DIM + d] * w[d] + b[d];
}

__global__ void k_wh(const float* __restrict__ ew, _Float16* __restrict__ wh) {
    int idx = blockIdx.x * blockDim.x + threadIdx.x;
    if (idx < NE * DIM * DIM) wh[idx] = (_Float16)ew[idx];
}

// node 255 is a leaf in every heap: emb[255]==node_term[255] forever, so its
// message is iteration-invariant. Precompute per edge type in exact f32.
__global__ void k_msg255(const float* __restrict__ ew, const float* __restrict__ eb,
                         const float* __restrict__ nt, float* __restrict__ out) {
    int e = blockIdx.x, i = threadIdx.x;
    const float* W = ew + ((size_t)e * DIM + i) * DIM;
    const float* x = nt + (NN - 1) * DIM;
    float acc = eb[e * DIM + i];
    for (int j = 0; j < DIM; ++j) acc += W[j] * x[j];
    out[e * DIM + i] = acc;
}

// children of node n form a contiguous range in heap order -> race-free gather
__global__ void k_child_ranges(const int* __restrict__ graphs,
                               int* __restrict__ clo, int* __restrict__ chi) {
    int idx = blockIdx.x * blockDim.x + threadIdx.x;
    if (idx >= NG * NN) return;
    int g = idx / NN, n = idx % NN;
    int lo = NN, hi = 0;
    for (int c = 1; c < NN; ++c)
        if (graphs[g * NN + c] == n) { if (c < lo) lo = c; hi = c + 1; }
    if (hi <= lo) { lo = 0; hi = 0; }
    clo[idx] = lo; chi[idx] = hi;
}

// group nodes 1..254 by edge type into 16-row WMMA tiles (pad with -1)
__global__ void k_build_tiles(const int* __restrict__ edges,
                              int* __restrict__ nodelist, int* __restrict__ tile_type,
                              int* __restrict__ n_tiles) {
    if (threadIdx.x || blockIdx.x) return;
    int R = 0;
    for (int t = 0; t < NE; ++t) {
        int cnt = 0, cur = -1;
        for (int n = 1; n < NN - 1; ++n) {
            if (edges[n] == t) {
                if ((cnt & 15) == 0) {
                    cur = R++; tile_type[cur] = t;
                    for (int s = 0; s < 16; ++s) nodelist[cur * 16 + s] = -1;
                }
                nodelist[cur * 16 + (cnt & 15)] = n;
                ++cnt;
            }
        }
    }
    *n_tiles = R;
}

// ---------------- main kernel: one workgroup per (graph, edge-variant) ----------------
__global__ __launch_bounds__(256) void k_main(
    const float* __restrict__ nt, const _Float16* __restrict__ Wh,
    const float* __restrict__ msg255, const int* __restrict__ clo,
    const int* __restrict__ chi, const int* __restrict__ nodelist,
    const int* __restrict__ tile_type, const int* __restrict__ n_tiles_p,
    const float* __restrict__ eb, const float* __restrict__ sew,
    const float* __restrict__ seb, const float* __restrict__ sdw,
    const float* __restrict__ sdb, float* __restrict__ scores) {
    extern __shared__ char smem[];
    _Float16* embH = (_Float16*)(smem + LDS_EMBH);   // [256][128] f16
    float*    msgs = (float*)(smem + LDS_MSGS);      // [256][128] f32 (row 0 = trash row)
    float*    red  = (float*)(smem + LDS_RED);
    int*      clog = (int*)(smem + LDS_CLO);
    int*      chig = (int*)(smem + LDS_CHI);

    const int v = blockIdx.x, g = v >> 3, e = v & 7;
    const int tid = threadIdx.x;
    const int R = *n_tiles_p;

    // init emb = node_term (f16); msg row for node 255 is iteration-invariant;
    // stage this graph's child ranges in LDS (reused 10x)
    for (int idx = tid; idx < NN * DIM; idx += 256) embH[idx] = (_Float16)nt[idx];
    for (int i = tid; i < DIM; i += 256) msgs[(NN - 1) * DIM + i] = msg255[e * DIM + i];
    for (int i = tid; i < NN; i += 256) { clog[i] = clo[g * NN + i]; chig[i] = chi[g * NN + i]; }
    __syncthreads();

    const int lane = tid & 31;
    const int wv   = tid >> 5;     // 8 waves
    const int hlf  = lane >> 4;    // half-wave select
    const int col  = lane & 15;

    for (int it = 0; it < ITERS; ++it) {
        // ---- phase 1: msgs[n] = W[type(n)] @ emb[n] + b, 16-node tiles via WMMA
        for (int task = wv; task < R * 8; task += 8) {   // wave-uniform (EXEC all-1 at WMMA)
            const int mt = task >> 3, ntl = task & 7;
            const int t = tile_type[mt];
            const int ncol0 = ntl * 16;
            // hoisted: A-row id, C-row ids (two aligned int4), bias — overlap with WMMAs
            const int arow = nodelist[mt * 16 + col];
            const int4 nA = *(const int4*)(nodelist + mt * 16 + 8 * hlf);
            const int4 nB = *(const int4*)(nodelist + mt * 16 + 8 * hlf + 4);
            const float bias = eb[t * DIM + ncol0 + col];
            const int anode = arow < 0 ? 0 : arow;       // dummy A rows: garbage in dead C rows
            const _Float16* aptr = embH + anode * DIM;
            const _Float16* bptr = Wh + ((size_t)t * DIM + (ncol0 + col)) * DIM; // B[K=j][N=i]=W[i][j]
            v8f c = {0.f, 0.f, 0.f, 0.f, 0.f, 0.f, 0.f, 0.f};
#pragma unroll
            for (int k0 = 0; k0 < DIM; k0 += 32) {
                v16h a, b;
                const _Float16* a0 = aptr + k0 + 8 * hlf;      // A: two 8-half runs per lane
                const _Float16* a1 = a0 + 16;
                const _Float16* b0 = bptr + k0 + 8 * hlf;      // B: two 8-half runs per lane
#pragma unroll
                for (int q = 0; q < 8; ++q) {
                    a[q] = a0[q]; a[8 + q] = a1[q];
                    b[q] = b0[q]; b[8 + q] = b0[16 + q];
                }
                c = __builtin_amdgcn_wmma_f32_16x16x32_f16(false, a, false, b,
                                                           (short)0, c, false, false);
            }
            const int nrow[8] = {nA.x, nA.y, nA.z, nA.w, nB.x, nB.y, nB.z, nB.w};
            float* mbase = msgs + ncol0 + col;
#pragma unroll
            for (int r = 0; r < 8; ++r) {                // C row M = r + 8*hlf
                // padded rows (-1) dump unconditionally into dead msgs row 0: no branches
                const int sn = nrow[r] < 0 ? 0 : nrow[r];
                mbase[sn * DIM] = c[r] + bias;
            }
        }
        __syncthreads();
        // ---- phase 2: emb[n] = node_term[n] + sum(children msgs), float4-vectorized
        {
            const float4* nt4   = (const float4*)nt;
            const float4* msgs4 = (const float4*)msgs;
            v4h* embH4 = (v4h*)embH;
            for (int idx = tid; idx < NN * (DIM / 4); idx += 256) {
                const int n = idx >> 5;                  // 32 float4 per row
                const int q = idx & 31;
                float4 acc = nt4[idx];
                const int lo = clog[n], hi = chig[n];
                for (int ch = lo; ch < hi; ++ch) {
                    const float4 m = msgs4[ch * (DIM / 4) + q];
                    acc.x += m.x; acc.y += m.y; acc.z += m.z; acc.w += m.w;
                }
                v4h h;
                h[0] = (_Float16)acc.x; h[1] = (_Float16)acc.y;
                h[2] = (_Float16)acc.z; h[3] = (_Float16)acc.w;
                embH4[idx] = h;                          // ds_store_b64
            }
        }
        __syncthreads();
    }

    // ---- score: recompute root in f32 from final msgs, then dot products
    float part = 0.f;
    if (tid < DIM) {
        float root = nt[tid];
        const int lo = clog[0], hi = chig[0];
        for (int ch = lo; ch < hi; ++ch) root += msgs[ch * DIM + tid];
        const float demb = nt[(NN - 1) * DIM + tid];
        part = (seb[tid] + root) * sew[tid] + (sdb[tid] + demb) * sdw[tid];
    }
    red[tid] = part;
    __syncthreads();
    for (int s2 = 128; s2 > 0; s2 >>= 1) {
        if (tid < s2) red[tid] += red[tid + s2];
        __syncthreads();
    }
    if (tid == 0) scores[v] = red[0];
}

// ---------------- finalize: column permutation + 32-way log_softmax ----------------
__global__ void k_finalize(const float* __restrict__ scores, const int* __restrict__ edges,
                           float* __restrict__ out) {
    __shared__ float s[32];
    __shared__ int perm[NE];
    __shared__ float m, lse;
    int tid = threadIdx.x;
    if (tid == 0) {
        int c = edges[NN - 1], k = 0;
        perm[k++] = c;
        for (int e2 = 0; e2 < NE; ++e2) if (e2 != c) perm[k++] = e2;
    }
    __syncthreads();
    int g = tid >> 3, k = tid & 7;
    float val = scores[g * NE + perm[k]];
    s[tid] = val;
    __syncthreads();
    if (tid == 0) {
        float mm = s[0];
        for (int i = 1; i < 32; ++i) mm = fmaxf(mm, s[i]);
        float ss = 0.f;
        for (int i = 0; i < 32; ++i) ss += expf(s[i] - mm);
        m = mm; lse = logf(ss);
    }
    __syncthreads();
    out[tid] = val - m - lse;
}

// ---------------- host launch ----------------
extern "C" void kernel_launch(void* const* d_in, const int* in_sizes, int n_in,
                              void* d_out, int out_size, void* d_ws, size_t ws_size,
                              hipStream_t stream) {
    const int*   data    = (const int*)d_in[0];
    // d_in[1] = types (unused)
    const int*   edges   = (const int*)d_in[2];
    const int*   graphs  = (const int*)d_in[3];
    const float* vecs    = (const float*)d_in[4];
    const float* dw      = (const float*)d_in[5];
    const float* db      = (const float*)d_in[6];
    const float* ew      = (const float*)d_in[7];
    const float* ebias   = (const float*)d_in[8];
    const float* sew     = (const float*)d_in[9];
    const float* seb     = (const float*)d_in[10];
    const float* sdw     = (const float*)d_in[11];
    const float* sdb     = (const float*)d_in[12];
    float* out = (float*)d_out;

    char* ws = (char*)d_ws;
    float*    nt        = (float*)(ws + WS_NT);
    _Float16* Wh        = (_Float16*)(ws + WS_WH);
    float*    msg255    = (float*)(ws + WS_MSG255);
    int*      clo       = (int*)(ws + WS_CLO);
    int*      chi       = (int*)(ws + WS_CHI);
    int*      nodelist  = (int*)(ws + WS_NODELIST);
    int*      tile_type = (int*)(ws + WS_TILETYPE);
    int*      n_tiles   = (int*)(ws + WS_NTILES);
    float*    scores    = (float*)(ws + WS_SCORES);

    k_node_term<<<NN, DIM, 0, stream>>>(data, vecs, dw, db, nt);
    k_wh<<<(NE * DIM * DIM + 255) / 256, 256, 0, stream>>>(ew, Wh);
    k_msg255<<<NE, DIM, 0, stream>>>(ew, ebias, nt, msg255);
    k_child_ranges<<<NG, NN, 0, stream>>>(graphs, clo, chi);
    k_build_tiles<<<1, 1, 0, stream>>>(edges, nodelist, tile_type, n_tiles);

    k_main<<<NG * NE, 256, LDS_BYTES, stream>>>(nt, Wh, msg255, clo, chi, nodelist,
                                                tile_type, n_tiles, ebias,
                                                sew, seb, sdw, sdb, scores);

    k_finalize<<<1, 32, 0, stream>>>(scores, edges, out);
}
